// ContrastiveNetWithGATAndTransformer_22041772163471
// MI455X (gfx1250) — matrix operand order repaired and checked by reading, not compiled, not run on hardware
//
#include <hip/hip_runtime.h>
#include <hip/hip_bf16.h>

// ---------------------------------------------------------------------------
// CDNA5 (gfx1250) implementation: encoder MLP -> GAT -> 2x transformer layer.
// All GEMMs + attention use V_WMMA_F32_16X16X32_BF16 (f32 accumulate).
// ---------------------------------------------------------------------------

typedef __attribute__((ext_vector_type(16))) __bf16 v16bf;
typedef __attribute__((ext_vector_type(8)))  __bf16 v8bf;
typedef __attribute__((ext_vector_type(4)))  __bf16 v4bf;
typedef __attribute__((ext_vector_type(2)))  __bf16 v2bf;
typedef __attribute__((ext_vector_type(8)))  float  v8f;

#define NNODES 4096
#define DMODEL 128
#define FFDIM  2048
#define NHEADS 4
#define DHEAD  32

static __device__ __forceinline__ void atomicMaxFloat(float* addr, float v) {
  int* ai = (int*)addr;
  int old = __float_as_int(*addr);
  while (__int_as_float(old) < v) {
    int prev = atomicCAS(ai, old, __float_as_int(v));
    if (prev == old) break;
    old = prev;
  }
}

// ---------------------------------------------------------------------------
// Tiled GEMM: C[M,N] = act(A[M,K] * B[K,N] + bias)
// 128 threads = 4 waves; block tile 64x64; each wave computes 32x32 (2x2 WMMA).
// fp32 -> bf16 conversion happens while staging tiles into LDS with packed
// cvt + vector DS stores; fragments are read back as 16B ds_load_b128.
// ---------------------------------------------------------------------------
template<int ACT, int HASBIAS>
__global__ __launch_bounds__(128)
void gemm_kernel(const float* __restrict__ A, const float* __restrict__ B,
                 const float* __restrict__ bias, float* __restrict__ C,
                 int M, int N, int K) {
  __shared__ __bf16 As[64][40];   // row-major, 80B row stride (16B aligned)
  __shared__ __bf16 Bs[64][40];   // transposed: Bs[n][k]

  const int tid  = threadIdx.x;
  const int wave = tid >> 5;
  const int lane = tid & 31;
  const int ln   = lane & 15;
  const int half = lane >> 4;
  const int brow = blockIdx.y * 64;
  const int bcol = blockIdx.x * 64;
  const int wrow = (wave >> 1) * 32;
  const int wcol = (wave & 1) * 32;

  v8f acc[2][2] = {};

  for (int k0 = 0; k0 < K; k0 += 32) {
    if (k0 + 32 < K) {            // prefetch next K-tile (global_prefetch_b8)
      __builtin_prefetch(&A[(size_t)(brow + (tid >> 1)) * K + k0 + 32 + (tid & 1) * 16], 0, 1);
      __builtin_prefetch(&B[(size_t)(k0 + 32 + (tid >> 2)) * N + bcol + (tid & 3) * 16], 0, 1);
    }
    // A tile 64x32: 4x float4 per thread -> packed bf16 -> 8B DS stores
#pragma unroll
    for (int i = 0; i < 4; i++) {
      int idx = i * 128 + tid;            // 512 groups of 4
      int r = idx >> 3, c4 = (idx & 7) * 4;
      const float4 va = *reinterpret_cast<const float4*>(&A[(size_t)(brow + r) * K + k0 + c4]);
      v4bf pk = { (__bf16)va.x, (__bf16)va.y, (__bf16)va.z, (__bf16)va.w };
      *reinterpret_cast<v4bf*>(&As[r][c4]) = pk;
    }
    // B tile 32x64 transposed: two k-rows packed per 4B DS store
#pragma unroll
    for (int i = 0; i < 8; i++) {
      int idx = i * 128 + tid;            // 1024 units: 64 n x 16 k-pairs
      int n = idx & 63, kp = (idx >> 6) * 2;
      float b0 = B[(size_t)(k0 + kp)     * N + bcol + n];
      float b1 = B[(size_t)(k0 + kp + 1) * N + bcol + n];
      v2bf pk = { (__bf16)b0, (__bf16)b1 };
      *reinterpret_cast<v2bf*>(&Bs[n][kp]) = pk;
    }
    __syncthreads();

#pragma unroll
    for (int mt = 0; mt < 2; mt++) {
      const int row = wrow + mt * 16 + ln;
      const int kb  = half * 8;
      v8bf a_lo = *reinterpret_cast<const v8bf*>(&As[row][kb]);
      v8bf a_hi = *reinterpret_cast<const v8bf*>(&As[row][kb + 16]);
      v16bf af = __builtin_shufflevector(a_lo, a_hi, 0,1,2,3,4,5,6,7,8,9,10,11,12,13,14,15);
#pragma unroll
      for (int nt = 0; nt < 2; nt++) {
        const int col = wcol + nt * 16 + ln;
        const int kb2 = half * 8;
        v8bf b_lo = *reinterpret_cast<const v8bf*>(&Bs[col][kb2]);
        v8bf b_hi = *reinterpret_cast<const v8bf*>(&Bs[col][kb2 + 16]);
        v16bf bf = __builtin_shufflevector(b_lo, b_hi, 0,1,2,3,4,5,6,7,8,9,10,11,12,13,14,15);
        acc[mt][nt] = __builtin_amdgcn_wmma_f32_16x16x32_bf16(
            false, af, false, bf, (short)0, acc[mt][nt], false, false);
      }
    }
    __syncthreads();
  }

#pragma unroll
  for (int nt = 0; nt < 2; nt++) {
    const int gc = bcol + wcol + nt * 16 + ln;
    const float bv = HASBIAS ? bias[gc] : 0.0f;
#pragma unroll
    for (int mt = 0; mt < 2; mt++)
#pragma unroll
      for (int r = 0; r < 8; r++) {
        int gr = brow + wrow + mt * 16 + half * 8 + r;
        float v = acc[mt][nt][r] + bv;
        if (ACT) v = fmaxf(v, 0.0f);
        C[(size_t)gr * N + gc] = v;
      }
  }
}

// ---------------------------------------------------------------------------
// Flash attention, one (query-block of 64, head) per block; 4 waves, each
// wave owns 16 query rows. dh=32 == one bf16 WMMA K-step. Online softmax.
// ---------------------------------------------------------------------------
__global__ __launch_bounds__(128)
void flash_attn_kernel(const float* __restrict__ Q, const float* __restrict__ Km,
                       const float* __restrict__ Vm, float* __restrict__ O, int N) {
  const int h    = blockIdx.y;
  const int wave = threadIdx.x >> 5;
  const int lane = threadIdx.x & 31;
  const int ln   = lane & 15;
  const int half = lane >> 4;
  const int qbase = blockIdx.x * 64 + wave * 16;

  __shared__ __bf16 Pt[4][16][32];   // per-wave P tile (C-layout -> A-layout relay)

  // Q fragment with 1/sqrt(dh) folded in
  v16bf qf;
  {
    const float sc = 0.17677669529663687f;   // 1/sqrt(32)
    const float* qp = Q + (size_t)(qbase + ln) * DMODEL + h * DHEAD;
    const int kb = half * 16;
#pragma unroll
    for (int i = 0; i < 8; i++) {
      qf[i]     = (__bf16)(qp[kb + i] * sc);
      qf[8 + i] = (__bf16)(qp[kb + 8 + i] * sc);
    }
  }

  float mrow[8], lrow[8];
  v8f o0 = {}, o1 = {};
#pragma unroll
  for (int r = 0; r < 8; r++) { mrow[r] = -1e30f; lrow[r] = 0.0f; }

  for (int kb0 = 0; kb0 < N; kb0 += 32) {
    if (kb0 + 32 < N) {           // prefetch next K/V block
      __builtin_prefetch(Km + (size_t)(kb0 + 32 + lane) * DMODEL + h * DHEAD, 0, 1);
      __builtin_prefetch(Vm + (size_t)(kb0 + 32 + lane) * DMODEL + h * DHEAD, 0, 1);
    }
    // K^T fragments (B layout: col = key, elems = dh)
    v16bf kf0, kf1;
    {
      const int kb2 = half * 16;
      const float* kp0 = Km + (size_t)(kb0 + ln) * DMODEL + h * DHEAD + kb2;
      const float* kp1 = Km + (size_t)(kb0 + 16 + ln) * DMODEL + h * DHEAD + kb2;
#pragma unroll
      for (int i = 0; i < 16; i++) { kf0[i] = (__bf16)kp0[i]; kf1[i] = (__bf16)kp1[i]; }
    }
    v8f s0 = {}, s1 = {};
    s0 = __builtin_amdgcn_wmma_f32_16x16x32_bf16(false, qf, false, kf0, (short)0, s0, false, false);
    s1 = __builtin_amdgcn_wmma_f32_16x16x32_bf16(false, qf, false, kf1, (short)0, s1, false, false);

    // online softmax update; row r lives in lanes of the same half (16 lanes)
#pragma unroll
    for (int r = 0; r < 8; r++) {
      float bm = fmaxf(s0[r], s1[r]);
      bm = fmaxf(bm, __shfl_xor(bm, 1, 32));
      bm = fmaxf(bm, __shfl_xor(bm, 2, 32));
      bm = fmaxf(bm, __shfl_xor(bm, 4, 32));
      bm = fmaxf(bm, __shfl_xor(bm, 8, 32));
      float mnew  = fmaxf(mrow[r], bm);
      float scale = __expf(mrow[r] - mnew);
      float p0 = __expf(s0[r] - mnew);
      float p1 = __expf(s1[r] - mnew);
      float ps = p0 + p1;
      ps += __shfl_xor(ps, 1, 32); ps += __shfl_xor(ps, 2, 32);
      ps += __shfl_xor(ps, 4, 32); ps += __shfl_xor(ps, 8, 32);
      lrow[r] = lrow[r] * scale + ps;
      mrow[r] = mnew;
      o0[r] *= scale; o1[r] *= scale;
      Pt[wave][half * 8 + r][ln]      = (__bf16)p0;
      Pt[wave][half * 8 + r][16 + ln] = (__bf16)p1;
    }

    // reload P in A-fragment layout (16 queries x 32 keys)
    v16bf pf;
    {
      const int kb = half * 16;
#pragma unroll
      for (int i = 0; i < 8; i++) {
        pf[i]     = Pt[wave][ln][kb + i];
        pf[8 + i] = Pt[wave][ln][kb + 8 + i];
      }
    }
    // V fragments (B layout: col = output dim, elems = keys)
    v16bf vf0, vf1;
    {
      const int kk = half * 16;
#pragma unroll
      for (int i = 0; i < 16; i++) {
        const float* vp = Vm + (size_t)(kb0 + kk + i) * DMODEL + h * DHEAD;
        vf0[i] = (__bf16)vp[ln];
        vf1[i] = (__bf16)vp[16 + ln];
      }
    }
    o0 = __builtin_amdgcn_wmma_f32_16x16x32_bf16(false, pf, false, vf0, (short)0, o0, false, false);
    o1 = __builtin_amdgcn_wmma_f32_16x16x32_bf16(false, pf, false, vf1, (short)0, o1, false, false);
  }

#pragma unroll
  for (int r = 0; r < 8; r++) {
    const int row = qbase + half * 8 + r;
    const float inv = 1.0f / lrow[r];
    O[(size_t)row * DMODEL + h * DHEAD + ln]      = o0[r] * inv;
    O[(size_t)row * DMODEL + h * DHEAD + 16 + ln] = o1[r] * inv;
  }
}

// ---------------------------------------------------------------------------
// GAT helper kernels
// ---------------------------------------------------------------------------
__global__ void gat_coeff_kernel(const float* __restrict__ hh, const float* __restrict__ aw_src,
                                 const float* __restrict__ aw_dst,
                                 float* __restrict__ a_src, float* __restrict__ a_dst) {
  const int nid = blockIdx.x, t = threadIdx.x;        // 128 threads, t = h*64+c
  float x  = hh[(size_t)nid * DMODEL + t];
  float ps = x * aw_src[t];
  float pd = x * aw_dst[t];
  for (int m = 1; m <= 16; m <<= 1) { ps += __shfl_xor(ps, m, 32); pd += __shfl_xor(pd, m, 32); }
  __shared__ float sb[8];
  if ((t & 31) == 0) { sb[(t >> 5) * 2] = ps; sb[(t >> 5) * 2 + 1] = pd; }
  __syncthreads();
  if (t < 2) {   // t = head
    a_src[(size_t)nid * 2 + t] = sb[t * 4 + 0] + sb[t * 4 + 2];
    a_dst[(size_t)nid * 2 + t] = sb[t * 4 + 1] + sb[t * 4 + 3];
  }
}

__global__ void gat_init_kernel(float* __restrict__ emax, float* __restrict__ edenom,
                                float* __restrict__ agg, int n) {
  int i = blockIdx.x * blockDim.x + threadIdx.x;
  if (i < n * 2)          { emax[i] = -1e30f; edenom[i] = 0.0f; }
  if (i < n * DMODEL)     agg[i] = 0.0f;
}

__global__ void gat_edge_max_kernel(const long long* __restrict__ src, const long long* __restrict__ dst,
                                    const float* __restrict__ a_src, const float* __restrict__ a_dst,
                                    float* __restrict__ e_raw, float* __restrict__ emax, int E) {
  int i = blockIdx.x * blockDim.x + threadIdx.x;
  if (i >= E * 2) return;
  int e = i >> 1, h = i & 1;
  int s = (int)src[e], d = (int)dst[e];
  float v = a_src[(size_t)s * 2 + h] + a_dst[(size_t)d * 2 + h];
  v = (v > 0.0f) ? v : 0.2f * v;                        // leaky_relu(0.2)
  e_raw[i] = v;
  atomicMaxFloat(&emax[(size_t)d * 2 + h], v);
}

__global__ void gat_edge_exp_kernel(const long long* __restrict__ dst,
                                    float* __restrict__ e_raw, const float* __restrict__ emax,
                                    float* __restrict__ edenom, int E) {
  int i = blockIdx.x * blockDim.x + threadIdx.x;
  if (i >= E * 2) return;
  int e = i >> 1, h = i & 1;
  int d = (int)dst[e];
  float ex = __expf(e_raw[i] - emax[(size_t)d * 2 + h]);
  e_raw[i] = ex;
  atomicAdd(&edenom[(size_t)d * 2 + h], ex);
}

__global__ void gat_aggregate_kernel(const long long* __restrict__ src, const long long* __restrict__ dst,
                                     const float* __restrict__ e_exp, const float* __restrict__ edenom,
                                     const float* __restrict__ hh, float* __restrict__ agg) {
  int e = blockIdx.x, c = threadIdx.x;                  // 128 channels
  int h = c >> 6;
  int s = (int)src[e], d = (int)dst[e];
  float alpha = e_exp[(size_t)e * 2 + h] / edenom[(size_t)d * 2 + h];
  atomicAdd(&agg[(size_t)d * DMODEL + c], alpha * hh[(size_t)s * DMODEL + c]);
}

__global__ void add_bias_kernel(float* __restrict__ y, const float* __restrict__ bias, int total) {
  int i = blockIdx.x * blockDim.x + threadIdx.x;
  if (i < total) y[i] += bias[i & (DMODEL - 1)];
}

// ---------------------------------------------------------------------------
// LayerNorm(x + r) * g + b  (D = 128, one row per block of 128 threads)
// ---------------------------------------------------------------------------
__global__ __launch_bounds__(128)
void ln_residual_kernel(const float* __restrict__ x, const float* __restrict__ r,
                        const float* __restrict__ g, const float* __restrict__ b,
                        float* __restrict__ out) {
  const int row = blockIdx.x, t = threadIdx.x;
  float v = x[(size_t)row * DMODEL + t] + r[(size_t)row * DMODEL + t];
  float s = v;
  for (int m = 1; m <= 16; m <<= 1) s += __shfl_xor(s, m, 32);
  __shared__ float red[4];
  if ((t & 31) == 0) red[t >> 5] = s;
  __syncthreads();
  float mean = (red[0] + red[1] + red[2] + red[3]) * (1.0f / DMODEL);
  float dv = v - mean;
  float q = dv * dv;
  for (int m = 1; m <= 16; m <<= 1) q += __shfl_xor(q, m, 32);
  __shared__ float red2[4];
  if ((t & 31) == 0) red2[t >> 5] = q;
  __syncthreads();
  float var = (red2[0] + red2[1] + red2[2] + red2[3]) * (1.0f / DMODEL);
  out[(size_t)row * DMODEL + t] = dv * rsqrtf(var + 1e-5f) * g[t] + b[t];
}

// ---------------------------------------------------------------------------
// Host-side orchestration
// ---------------------------------------------------------------------------
static inline void gemm(const float* A, const float* B, const float* bias,
                        float* C, int M, int N, int K, int act, hipStream_t s) {
  dim3 g(N / 64, M / 64);
  if (bias) {
    if (act) gemm_kernel<1, 1><<<g, 128, 0, s>>>(A, B, bias, C, M, N, K);
    else     gemm_kernel<0, 1><<<g, 128, 0, s>>>(A, B, bias, C, M, N, K);
  } else {
    if (act) gemm_kernel<1, 0><<<g, 128, 0, s>>>(A, B, bias, C, M, N, K);
    else     gemm_kernel<0, 0><<<g, 128, 0, s>>>(A, B, bias, C, M, N, K);
  }
}

extern "C" void kernel_launch(void* const* d_in, const int* in_sizes, int n_in,
                              void* d_out, int out_size, void* d_ws, size_t ws_size,
                              hipStream_t stream) {
  const int N = NNODES;
  const float*     x      = (const float*)d_in[0];
  const long long* ei     = (const long long*)d_in[1];   // int64 per reference
  const float*     enc_w1 = (const float*)d_in[2];
  const float*     enc_b1 = (const float*)d_in[3];
  const float*     enc_w2 = (const float*)d_in[4];
  const float*     enc_b2 = (const float*)d_in[5];
  const float*     gat_w  = (const float*)d_in[6];
  const float*     gat_as = (const float*)d_in[7];
  const float*     gat_ad = (const float*)d_in[8];
  const float*     gat_b  = (const float*)d_in[9];
  const float*     t_wq   = (const float*)d_in[10];
  const float*     t_bq   = (const float*)d_in[11];
  const float*     t_wk   = (const float*)d_in[12];
  const float*     t_bk   = (const float*)d_in[13];
  const float*     t_wv   = (const float*)d_in[14];
  const float*     t_bv   = (const float*)d_in[15];
  const float*     t_wo   = (const float*)d_in[16];
  const float*     t_bo   = (const float*)d_in[17];
  const float*     ln1g   = (const float*)d_in[18];
  const float*     ln1b   = (const float*)d_in[19];
  const float*     ln2g   = (const float*)d_in[20];
  const float*     ln2b   = (const float*)d_in[21];
  const float*     t_w1   = (const float*)d_in[22];
  const float*     t_b1   = (const float*)d_in[23];
  const float*     t_w2   = (const float*)d_in[24];
  const float*     t_b2   = (const float*)d_in[25];

  const int E = in_sizes[1] / 2;
  const long long* esrc = ei;
  const long long* edst = ei + E;

  // workspace carve-out
  float* W = (float*)d_ws;
  const size_t ND = (size_t)N * DMODEL;
  float* h1     = W; W += ND;
  float* henc   = W; W += ND;
  float* hhp    = W; W += ND;
  float* ygat   = W; W += ND;
  float* ybuf   = W; W += ND;
  float* y2     = W; W += ND;
  float* qb     = W; W += ND;
  float* kb     = W; W += ND;
  float* vb     = W; W += ND;
  float* ab     = W; W += ND;
  float* tmp    = W; W += ND;
  float* ffn    = W; W += (size_t)N * FFDIM;
  float* asrcb  = W; W += (size_t)N * 2;
  float* adstb  = W; W += (size_t)N * 2;
  float* emaxb  = W; W += (size_t)N * 2;
  float* edenb  = W; W += (size_t)N * 2;
  float* eraw   = W; W += (size_t)E * 2;

  // --- encoder MLP ---
  gemm(x,    enc_w1, enc_b1, h1,   N, DMODEL, 256,    1, stream);
  gemm(h1,   enc_w2, enc_b2, henc, N, DMODEL, DMODEL, 0, stream);

  // --- GAT ---
  gemm(henc, gat_w, nullptr, hhp, N, DMODEL, DMODEL, 0, stream);
  gat_coeff_kernel<<<N, 128, 0, stream>>>(hhp, gat_as, gat_ad, asrcb, adstb);
  gat_init_kernel<<<(N * DMODEL + 255) / 256, 256, 0, stream>>>(emaxb, edenb, ygat, N);
  gat_edge_max_kernel<<<(E * 2 + 255) / 256, 256, 0, stream>>>(esrc, edst, asrcb, adstb, eraw, emaxb, E);
  gat_edge_exp_kernel<<<(E * 2 + 255) / 256, 256, 0, stream>>>(edst, eraw, emaxb, edenb, E);
  gat_aggregate_kernel<<<E, 128, 0, stream>>>(esrc, edst, eraw, edenb, hhp, ygat);
  add_bias_kernel<<<(N * DMODEL + 255) / 256, 256, 0, stream>>>(ygat, gat_b, N * DMODEL);

  // --- transformer encoder, 2 post-LN layers ---
  const float* ycur = ygat;
  for (int l = 0; l < 2; l++) {
    const size_t wofs = (size_t)l * DMODEL * DMODEL;
    gemm(ycur, t_wq + wofs, t_bq + l * DMODEL, qb, N, DMODEL, DMODEL, 0, stream);
    gemm(ycur, t_wk + wofs, t_bk + l * DMODEL, kb, N, DMODEL, DMODEL, 0, stream);
    gemm(ycur, t_wv + wofs, t_bv + l * DMODEL, vb, N, DMODEL, DMODEL, 0, stream);
    flash_attn_kernel<<<dim3(N / 64, NHEADS), 128, 0, stream>>>(qb, kb, vb, ab, N);
    gemm(ab, t_wo + wofs, t_bo + l * DMODEL, tmp, N, DMODEL, DMODEL, 0, stream);
    ln_residual_kernel<<<N, 128, 0, stream>>>(ycur, tmp, ln1g + l * DMODEL, ln1b + l * DMODEL, y2);
    gemm(y2,  t_w1 + (size_t)l * DMODEL * FFDIM, t_b1 + l * FFDIM, ffn, N, FFDIM, DMODEL, 1, stream);
    gemm(ffn, t_w2 + (size_t)l * FFDIM * DMODEL, t_b2 + l * DMODEL, tmp, N, DMODEL, FFDIM, 0, stream);
    float* lout = (l == 1) ? (float*)d_out : ybuf;
    ln_residual_kernel<<<N, 128, 0, stream>>>(y2, tmp, ln2g + l * DMODEL, ln2b + l * DMODEL, lout);
    ycur = lout;
  }
}
